// NeuraMatch_82240033783847
// MI455X (gfx1250) — matrix operand order repaired and checked by reading, not compiled, not run on hardware
//
#include <hip/hip_runtime.h>
#include <hip/hip_bf16.h>
#include <math.h>
#include <stdint.h>

typedef __attribute__((ext_vector_type(16))) _Float16 v16h;
typedef __attribute__((ext_vector_type(8)))  float    v8f;

#define AS1 __attribute__((address_space(1)))
#define AS3 __attribute__((address_space(3)))

#if __has_builtin(__builtin_amdgcn_global_load_async_to_lds_b32)
#define HAS_ASYNC_LDS 1
#else
#define HAS_ASYNC_LDS 0
#endif

__device__ __forceinline__ void wait_async0() {
#if __has_builtin(__builtin_amdgcn_s_wait_asynccnt)
  __builtin_amdgcn_s_wait_asynccnt(0);
#elif HAS_ASYNC_LDS
  asm volatile("s_wait_asynccnt 0x0" ::: "memory");
#endif
}

// ---------------------------------------------------------------------------
// Implicit-GEMM conv / conv-transpose with CDNA5 WMMA (wave32, 16x16x32 f16).
// Templated on (K, S, TR): all window-decode divisions are by constants.
// One wave computes TWO 16(M=Cout) x 16(N=pixels) f32 accumulator tiles
// sharing one A (weight) fragment. Weights are staged fp32 into a
// DOUBLE-BUFFERED LDS tile with GLOBAL_LOAD_ASYNC_TO_LDS (ASYNCcnt),
// overlapping the DMA of chunk ch+1 with the gather+WMMA of chunk ch.
// ---------------------------------------------------------------------------
template <int K, int S, int TR>
__global__ __launch_bounds__(128) void conv_wmma(
    const float* __restrict__ in, const float* __restrict__ w,
    const float* __restrict__ bias, float* __restrict__ out,
    int Nimg, int Cin, int Hin, int Win, int Cout, int Hout, int Wout,
    int concat) {
  constexpr int KK = K * K;
  __shared__ float ldsW[2][16 * 32];          // fp32 weight chunk, M x Kc

  const int lane  = threadIdx.x & 31;
  const int wv    = threadIdx.x >> 5;
  const int mtile = blockIdx.y;

  const unsigned HW   = (unsigned)(Hout * Wout);
  const unsigned Ntot = (unsigned)Nimg * HW;
  const int Ktot   = Cin * KK;
  const int nchunk = (Ktot + 31) >> 5;

  // two N tiles per wave
  const unsigned tile0 = ((unsigned)blockIdx.x * 4u + (unsigned)wv) * 2u;
  const int ncol = lane & 15;
  unsigned np0 = tile0 * 16u + (unsigned)ncol;
  unsigned np1 = np0 + 16u;
  const bool ok0 = (np0 < Ntot), ok1 = (np1 < Ntot);
  if (!ok0) np0 = Ntot - 1u;
  if (!ok1) np1 = Ntot - 1u;
  const int img0 = (int)(np0 / HW); const unsigned r0 = np0 % HW;
  const int oy0  = (int)(r0 / (unsigned)Wout), ox0 = (int)(r0 % (unsigned)Wout);
  const int img1 = (int)(np1 / HW); const unsigned r1 = np1 % HW;
  const int oy1  = (int)(r1 / (unsigned)Wout), ox1 = (int)(r1 % (unsigned)Wout);

  // per-lane fragment bases: contiguous k = 16*(lane/16) + e, same order for A and B
  const int kb_b = (lane & 16) ? 16 : 0;   // B: K base
  const int m_a  = lane & 15;              // A: row
  const int kb_a = (lane & 16) ? 16 : 0;   // A: K sub-base

  // weight chunk staging: row m of the A tile is contiguous in memory
  // (w + co*Ktot + kbase), no index decode needed.
  auto stage = [&](int ch, float* buf) {
    const int kbase = ch * 32;
    for (int t = threadIdx.x; t < 512; t += 128) {   // 4 iters per thread
      const int m  = t >> 5;
      const int kk = t & 31;
      const int co = mtile * 16 + m;
      const int kg = kbase + kk;
      if (co < Cout && kg < Ktot) {
#if HAS_ASYNC_LDS
        __builtin_amdgcn_global_load_async_to_lds_b32(
            (AS1 int*)(const_cast<float*>(w) + co * Ktot + kg),
            (AS3 int*)(buf + t), 0, 0);
#else
        buf[t] = w[co * Ktot + kg];
#endif
      } else {
        buf[t] = 0.f;
      }
    }
  };

  v8f acc0 = {}, acc1 = {};

  stage(0, ldsW[0]);
  wait_async0();
  __syncthreads();

  for (int ch = 0; ch < nchunk; ++ch) {
    const int kbase = ch * 32;
    const float* cur = ldsW[ch & 1];

    // prefetch-stage next chunk into the other LDS buffer (async DMA)
    if (ch + 1 < nchunk) stage(ch + 1, ldsW[(ch + 1) & 1]);

    // ---- A fragment from LDS (16-bit A 16x32 layout), f32 -> f16 ----
    v16h af;
#pragma unroll
    for (int e = 0; e < 16; ++e) {
      const int k = kb_a + e;
      af[e] = (_Float16)cur[m_a * 32 + k];
    }

    // prefetch one K-chunk of activations ahead
    if (ch + 1 < nchunk) {
      const int kg = kbase + 32 + kb_b;
      if (kg < Ktot) {
        const int ci = kg / KK;
        __builtin_prefetch(&in[(img0 * Cin + ci) * Hin * Win], 0, 1);
      }
    }

    // ---- B fragments gathered from global (32x16 K x N layout) ----
    auto gatherB = [&](int img, int oy, int ox) -> v16h {
      v16h bf;
#pragma unroll
      for (int e = 0; e < 16; ++e) {
        const int kg = kbase + kb_b + e;
        float v = 0.f;
        if (kg < Ktot) {
          const int ci = kg / KK, r2 = kg % KK;    // constant divisors
          const int ky = r2 / K,  kx = r2 % K;
          if constexpr (!TR) {
            v = in[((img * Cin + ci) * Hin + oy * S + ky) * Win + ox * S + kx];
          } else {
            const int ty = oy - ky, tx = ox - kx;
            if (ty >= 0 && tx >= 0 && (ty % S) == 0 && (tx % S) == 0) {
              const int iy = ty / S, ix = tx / S;  // S is 1 or 2: bit ops
              if (iy < Hin && ix < Win)
                v = in[((img * Cin + ci) * Hin + iy) * Win + ix];
            }
          }
        }
        bf[e] = (_Float16)v;
      }
      return bf;
    };
    const v16h bf0 = gatherB(img0, oy0, ox0);
    const v16h bf1 = gatherB(img1, oy1, ox1);

    // two WMMAs sharing the A fragment (uniform control flow: EXEC all-ones)
    acc0 = __builtin_amdgcn_wmma_f32_16x16x32_f16(false, af, false, bf0,
                                                  (short)0, acc0, false, false);
    acc1 = __builtin_amdgcn_wmma_f32_16x16x32_f16(false, af, false, bf1,
                                                  (short)0, acc1, false, false);

    wait_async0();          // next-chunk DMA complete (own wave's ASYNCcnt)
    __syncthreads();        // all waves done reading cur / writing next
  }

  // ---- store C tiles (VGPR r -> row r + 8*(lane>=16), col lane&15) ----
  const int mrow0 = (lane & 16) ? 8 : 0;
  auto store = [&](const v8f& acc, bool ok, int img, int oy, int ox) {
    if (!ok) return;
#pragma unroll
    for (int rr = 0; rr < 8; ++rr) {
      const int m  = mrow0 + rr;
      const int co = mtile * 16 + m;
      if (co < Cout) {
        float v = acc[rr];
        if (bias) v += bias[co];
        int off;
        if (!concat) {
          off = ((img * Cout + co) * Hout + oy) * Wout + ox;
        } else {                       // v_ab concat: [b][64][47][47]
          const int b   = img >> 1;
          const int chn = (img & 1) * 32 + co;
          off = ((b * 64 + chn) * Hout + oy) * Wout + ox;
        }
        out[off] = v;
      }
    }
  };
  store(acc0, ok0, img0, oy0, ox0);
  store(acc1, ok1, img1, oy1, ox1);
}

// ---------------------------------------------------------------------------
// Training-mode batch norm: per-channel mean / inv-std over (N,H,W)
// ---------------------------------------------------------------------------
__global__ __launch_bounds__(256) void bn_stats(const float* __restrict__ x,
                                                float* __restrict__ stats,
                                                int C, int Nimg, int HW) {
  __shared__ float s_sum[256], s_sq[256];
  const int c = blockIdx.x;
  float sum = 0.f, sq = 0.f;
  for (int n = 0; n < Nimg; ++n) {
    const float* p = x + (size_t)(n * C + c) * (size_t)HW;
    for (int r = threadIdx.x; r < HW; r += 256) {
      const float v = p[r];
      sum += v; sq += v * v;
    }
  }
  s_sum[threadIdx.x] = sum; s_sq[threadIdx.x] = sq;
  __syncthreads();
  for (int st = 128; st > 0; st >>= 1) {
    if (threadIdx.x < st) {
      s_sum[threadIdx.x] += s_sum[threadIdx.x + st];
      s_sq [threadIdx.x] += s_sq [threadIdx.x + st];
    }
    __syncthreads();
  }
  if (threadIdx.x == 0) {
    const float cnt = (float)Nimg * (float)HW;
    const float m   = s_sum[0] / cnt;
    const float var = s_sq[0] / cnt - m * m;
    stats[c * 2 + 0] = m;
    stats[c * 2 + 1] = rsqrtf(var + 1e-5f);
  }
}

// grid: (ceil(HW/256), Nimg*C); channel id from uniform blockIdx.y
__global__ void bn_apply_lrelu(float* __restrict__ x, const float* __restrict__ stats,
                               const float* __restrict__ g, const float* __restrict__ b,
                               int C, int HW) {
  const int c = (int)blockIdx.y % C;
  const int r = (int)blockIdx.x * 256 + (int)threadIdx.x;
  if (r >= HW) return;
  float* p = x + (size_t)blockIdx.y * (size_t)HW;
  const float mean = stats[c * 2], inv = stats[c * 2 + 1];
  float v = (p[r] - mean) * inv * g[c] + b[c];
  p[r] = (v > 0.f) ? v : 0.01f * v;
}

// mode 0 = relu, 1 = tanh
__global__ void act_copy(const float* __restrict__ in, float* __restrict__ out,
                         unsigned total, int mode) {
  const unsigned i = blockIdx.x * 256u + threadIdx.x;
  if (i >= total) return;
  const float v = in[i];
  out[i] = (mode == 0) ? fmaxf(v, 0.f) : tanhf(v);
}

// ---------------------------------------------------------------------------
// Match extraction: mv -> target coords -> gathered confidence -> mask
// ---------------------------------------------------------------------------
__global__ void postprocess(const float* __restrict__ hm, const float* __restrict__ mv,
                            float* __restrict__ conf_mask, float* __restrict__ mask,
                            float* __restrict__ targ, int nb) {
  constexpr int S = 256, SS = S * S;
  const unsigned i = blockIdx.x * 256u + threadIdx.x;
  const unsigned tot = (unsigned)nb * SS;
  if (i >= tot) return;
  const int b = (int)(i >> 16), pix = (int)(i & (SS - 1));
  const int px = pix & (S - 1), py = pix >> 8;
  const int mvx = (int)lrintf(mv[(b * 2 + 0) * SS + pix] * (float)(S - 1));
  const int mvy = (int)lrintf(mv[(b * 2 + 1) * SS + pix] * (float)(S - 1));
  const int tx = min(max(px + mvx, 0), S - 1);
  const int ty = min(max(py + mvy, 0), S - 1);
  const float conf_t = hm[(b * 2 + 1) * SS + ty * S + tx];
  const float cm = 0.5f * (hm[(b * 2 + 0) * SS + pix] + conf_t);
  conf_mask[b * SS + pix] = cm;
  mask[b * SS + pix]      = (cm > 0.5f) ? 1.f : 0.f;
  targ[(b * 2 + 0) * SS + pix] = (float)tx;
  targ[(b * 2 + 1) * SS + pix] = (float)ty;
}

// ---------------------------------------------------------------------------
// Host orchestration
// ---------------------------------------------------------------------------
static void launch_conv(hipStream_t stream, const float* in, const float* w,
                        const float* bias, float* out, int Nimg, int Cin, int Hin,
                        int Win, int Cout, int Hout, int Wout, int K, int S, int TR,
                        int concat) {
  const unsigned Ntot = (unsigned)Nimg * (unsigned)(Hout * Wout);
  const unsigned ntiles = (Ntot + 15u) / 16u;
  dim3 grid((ntiles + 7u) / 8u, (unsigned)((Cout + 15) / 16));
  dim3 blk(128);
#define CONV_CASE(k, s, t)                                                    \
  if (K == k && S == s && TR == t) {                                          \
    hipLaunchKernelGGL((conv_wmma<k, s, t>), grid, blk, 0, stream, in, w,     \
                       bias, out, Nimg, Cin, Hin, Win, Cout, Hout, Wout,      \
                       concat);                                               \
    return;                                                                   \
  }
  CONV_CASE(5, 2, 0)
  CONV_CASE(7, 1, 0)
  CONV_CASE(9, 1, 0)
  CONV_CASE(9, 1, 1)
  CONV_CASE(7, 1, 1)
  CONV_CASE(6, 2, 1)
#undef CONV_CASE
}

extern "C" void kernel_launch(void* const* d_in, const int* in_sizes, int n_in,
                              void* d_out, int out_size, void* d_ws, size_t ws_size,
                              hipStream_t stream) {
  (void)in_sizes; (void)n_in; (void)out_size; (void)ws_size;

  const float* x        = (const float*)d_in[0];
  const float* enc_w[4] = {(const float*)d_in[1], (const float*)d_in[2],
                           (const float*)d_in[3], (const float*)d_in[4]};
  const float* enc_g[3] = {(const float*)d_in[5], (const float*)d_in[7], (const float*)d_in[9]};
  const float* enc_b[3] = {(const float*)d_in[6], (const float*)d_in[8], (const float*)d_in[10]};
  const float* enc_bias = (const float*)d_in[11];
  const float* dw[2][4] = {{(const float*)d_in[12], (const float*)d_in[13],
                            (const float*)d_in[14], (const float*)d_in[15]},
                           {(const float*)d_in[23], (const float*)d_in[24],
                            (const float*)d_in[25], (const float*)d_in[26]}};
  const float* dg[2][3] = {{(const float*)d_in[16], (const float*)d_in[18], (const float*)d_in[20]},
                           {(const float*)d_in[27], (const float*)d_in[29], (const float*)d_in[31]}};
  const float* db[2][3] = {{(const float*)d_in[17], (const float*)d_in[19], (const float*)d_in[21]},
                           {(const float*)d_in[28], (const float*)d_in[30], (const float*)d_in[32]}};
  const float* dbias[2] = {(const float*)d_in[22], (const float*)d_in[33]};

  // workspace layout (floats): ping | pong | v_ab | bn stats
  float* ws    = (float*)d_ws;
  float* regA  = ws;                      // 16*32*126*126 = 8,128,512 max
  float* regB  = ws + 8128512;            // 16*64*61*61   = 3,810,304 max
  float* vab   = ws + 8128512 + 3810304;  // 8*64*47*47    = 1,131,008
  float* stats = vab + 1131008;           // 2 * 128

  float* o       = (float*)d_out;
  float* out_hm  = o;                     // [8,2,256,256]
  float* out_mv  = o + 1048576;           // [8,2,256,256]
  float* out_cm  = o + 2097152;           // [8,256,256]
  float* out_msk = o + 2621440;           // [8,256,256]
  float* out_tg  = o + 3145728;           // [8,2,65536]

  auto bn = [&](float* t, const float* g, const float* b, int C, int Nimg, int HW) {
    hipLaunchKernelGGL(bn_stats, dim3((unsigned)C), dim3(256), 0, stream,
                       (const float*)t, stats, C, Nimg, HW);
    dim3 grid((unsigned)((HW + 255) / 256), (unsigned)(Nimg * C));
    hipLaunchKernelGGL(bn_apply_lrelu, grid, dim3(256), 0, stream, t,
                       (const float*)stats, g, b, C, HW);
  };

  // ---------------- encoder on 16 images (both branches) ----------------
  launch_conv(stream, x,    enc_w[0], nullptr,  regA, 16,   3, 256, 256,  32, 126, 126, 5, 2, 0, 0);
  bn(regA, enc_g[0], enc_b[0], 32, 16, 126 * 126);
  launch_conv(stream, regA, enc_w[1], nullptr,  regB, 16,  32, 126, 126,  64,  61,  61, 5, 2, 0, 0);
  bn(regB, enc_g[1], enc_b[1], 64, 16, 61 * 61);
  launch_conv(stream, regB, enc_w[2], nullptr,  regA, 16,  64,  61,  61, 128,  55,  55, 7, 1, 0, 0);
  bn(regA, enc_g[2], enc_b[2], 128, 16, 55 * 55);
  launch_conv(stream, regA, enc_w[3], enc_bias, vab,  16, 128,  55,  55,  32,  47,  47, 9, 1, 0, 1);

  // ---------------- two decoders on v_ab [8,64,47,47] ----------------
  float* dfinal[2] = {out_hm, out_mv};
  const int dmode[2] = {0, 1};  // relu, tanh
  for (int d = 0; d < 2; ++d) {
    launch_conv(stream, vab,  dw[d][0], nullptr, regA, 8, 64, 47, 47, 64, 55, 55, 9, 1, 1, 0);
    bn(regA, dg[d][0], db[d][0], 64, 8, 55 * 55);
    launch_conv(stream, regA, dw[d][1], nullptr, regB, 8, 64, 55, 55, 32, 61, 61, 7, 1, 1, 0);
    bn(regB, dg[d][1], db[d][1], 32, 8, 61 * 61);
    launch_conv(stream, regB, dw[d][2], nullptr, regA, 8, 32, 61, 61, 16, 126, 126, 6, 2, 1, 0);
    bn(regA, dg[d][2], db[d][2], 16, 8, 126 * 126);
    launch_conv(stream, regA, dw[d][3], dbias[d], regB, 8, 16, 126, 126, 2, 256, 256, 6, 2, 1, 0);
    const unsigned tot = 8u * 2u * 256u * 256u;
    hipLaunchKernelGGL(act_copy, dim3((tot + 255u) / 256u), dim3(256), 0, stream,
                       (const float*)regB, dfinal[d], tot, dmode[d]);
  }

  // ---------------- match extraction ----------------
  const unsigned ptot = 8u * 256u * 256u;
  hipLaunchKernelGGL(postprocess, dim3((ptot + 255u) / 256u), dim3(256), 0, stream,
                     (const float*)out_hm, (const float*)out_mv,
                     out_cm, out_msk, out_tg, 8);
}
